// ChromatinGAT_82231443849544
// MI455X (gfx1250) — matrix operand order repaired and checked by reading, not compiled, not run on hardware
//
#include <hip/hip_runtime.h>
#include <hip/hip_bf16.h>

// ---------------------------------------------------------------------------
// ChromatinGAT on MI455X (gfx1250): 3x GATConv + BN/ELU + MLP head.
// Dense transforms use v_wmma_f32_16x16x32_f16 (f16 in, f32 accum),
// A fragments loaded as 2x b128 straight from row-major activations,
// B fragments from a pre-packed (fragment-ordered) f16 weight buffer.
// K-loop fully unrolled (templated) with loads hoisted ahead of the WMMAs.
// Edge softmax/aggregation is HBM-bound gather/scatter with f32 atomics.
// ---------------------------------------------------------------------------

typedef __attribute__((ext_vector_type(16))) _Float16 v16h;
typedef __attribute__((ext_vector_type(8)))  _Float16 v8h;
typedef __attribute__((ext_vector_type(4)))  _Float16 v4h;
typedef __attribute__((ext_vector_type(8)))  float    v8f;

#define NNODES 50000
#define NEDGES 800000
#define NETOT  (NEDGES + NNODES)

#define CAT16(lo, hi)                                                         \
  __builtin_shufflevector(lo, hi, 0, 1, 2, 3, 4, 5, 6, 7, 8, 9, 10, 11, 12,  \
                          13, 14, 15)

// ---------------- utility: ordered-float <-> uint (for atomic max) ---------
__device__ __forceinline__ unsigned f2o(float f) {
  unsigned u = __float_as_uint(f);
  return (u & 0x80000000u) ? ~u : (u | 0x80000000u);
}
__device__ __forceinline__ float o2f(unsigned u) {
  return (u & 0x80000000u) ? __uint_as_float(u & 0x7FFFFFFFu)
                           : __uint_as_float(~u);
}

// ---------------- f32 -> f16 cast (4 elements/thread) ----------------------
__global__ void cast_f16_kernel(const float* __restrict__ in,
                                _Float16* __restrict__ out, int n4) {
  int i = blockIdx.x * blockDim.x + threadIdx.x;
  if (i >= n4) return;
  float4 v = ((const float4*)in)[i];
  v4h o = { (_Float16)v.x, (_Float16)v.y, (_Float16)v.z, (_Float16)v.w };
  ((v4h*)out)[i] = o;
}

// ---------------- pack weights into WMMA B-fragment order ------------------
// Bp[((kt*tilesN + tn)*32 + lane)*16 + e] = (f16) W[(kt*32 + 8*half + (e&7) + 16*(e>>3))*Nc
//                                                   + tn*16 + (lane&15)]
__global__ void pack_w_kernel(const float* __restrict__ W,
                              _Float16* __restrict__ Bp, int K, int Nc) {
  int t = blockIdx.x * blockDim.x + threadIdx.x;
  int total = (K >> 5) * (Nc >> 4) * 32;
  if (t >= total) return;
  int lane = t & 31;
  int tile = t >> 5;
  int tilesN = Nc >> 4;
  int tn = tile % tilesN;
  int kt = tile / tilesN;
  int half = lane >> 4, l16 = lane & 15;
  _Float16* dst = Bp + ((size_t)t << 4);
  const float* src = W + (size_t)(kt * 32 + (half << 3)) * Nc + tn * 16 + l16;
#pragma unroll
  for (int e = 0; e < 16; ++e) dst[e] = (_Float16)src[(size_t)(e + ((e >> 3) << 3)) * Nc];
}

// ---------------- WMMA GEMM: C[M,Nc] = A[M,K] @ W[K,Nc] --------------------
// One wave computes 16 rows x 64 cols (4 tiles). K templated, fully unrolled;
// all 5 fragments of a k-step loaded before the 4 back-to-back WMMAs.
template <int K>
__global__ void gemm_wmma_kernel(const _Float16* __restrict__ A,
                                 const _Float16* __restrict__ Bp,
                                 float* __restrict__ C, int Nc) {
  const int tm   = blockIdx.x;            // 16-row tile
  const int tn0  = blockIdx.y << 2;       // first of 4 16-col tiles
  const int lane = threadIdx.x & 31;
  const int half = lane >> 4;
  const int l16  = lane & 15;
  const int tilesN = Nc >> 4;

  v8f acc0 = {}, acc1 = {}, acc2 = {}, acc3 = {};
  // A fragment: two contiguous 8-f16 runs per lane (ISA 7.12.2 layout)
  const _Float16* arow = A + (size_t)(tm * 16 + l16) * K + (half << 3);
  // B fragments: 16 contiguous f16 per lane, tiles 512 f16 apart
  const _Float16* bcol = Bp + (((size_t)tn0 * 32 + lane) << 4);
  const size_t bstride = (size_t)tilesN * 512;   // per 32-wide k-chunk

#pragma unroll
  for (int k0 = 0; k0 < K; k0 += 32) {
    v8h a0 = *(const v8h*)(arow + k0);
    v8h a1 = *(const v8h*)(arow + k0 + 16);
    const _Float16* bb = bcol + (size_t)(k0 >> 5) * bstride;
    v8h b0l = *(const v8h*)(bb);          v8h b0h = *(const v8h*)(bb + 8);
    v8h b1l = *(const v8h*)(bb + 512);    v8h b1h = *(const v8h*)(bb + 520);
    v8h b2l = *(const v8h*)(bb + 1024);   v8h b2h = *(const v8h*)(bb + 1032);
    v8h b3l = *(const v8h*)(bb + 1536);   v8h b3h = *(const v8h*)(bb + 1544);
    v16h a  = CAT16(a0, a1);
    v16h b0 = CAT16(b0l, b0h);
    v16h b1 = CAT16(b1l, b1h);
    v16h b2 = CAT16(b2l, b2h);
    v16h b3 = CAT16(b3l, b3h);
    acc0 = __builtin_amdgcn_wmma_f32_16x16x32_f16(false, a, false, b0,
                                                  (short)0, acc0, false, false);
    acc1 = __builtin_amdgcn_wmma_f32_16x16x32_f16(false, a, false, b1,
                                                  (short)0, acc1, false, false);
    acc2 = __builtin_amdgcn_wmma_f32_16x16x32_f16(false, a, false, b2,
                                                  (short)0, acc2, false, false);
    acc3 = __builtin_amdgcn_wmma_f32_16x16x32_f16(false, a, false, b3,
                                                  (short)0, acc3, false, false);
  }
#pragma unroll
  for (int t = 0; t < 4; ++t) {
    v8f c = (t == 0) ? acc0 : (t == 1) ? acc1 : (t == 2) ? acc2 : acc3;
    int col = (tn0 + t) * 16 + l16;
#pragma unroll
    for (int v = 0; v < 8; ++v) {
      int row = tm * 16 + v + (half << 3);   // C/D: M = v + 8*half
      C[(size_t)row * Nc + col] = c[v];
    }
  }
}

// ---------------- attention coefficients ----------------------------------
__global__ void att_coef_kernel(const float* __restrict__ h,
                                const float* __restrict__ as,
                                const float* __restrict__ ad,
                                float* __restrict__ asrc,
                                float* __restrict__ adst,
                                int n, int H, int F) {
  int i = blockIdx.x * blockDim.x + threadIdx.x;
  if (i >= n * H) return;
  int node = i / H, hd = i % H;
  const float* hp  = h + (size_t)node * H * F + (size_t)hd * F;
  const float* asp = as + hd * F;
  const float* adp = ad + hd * F;
  float s1 = 0.f, s2 = 0.f;
#pragma unroll 4
  for (int f = 0; f < F; ++f) {
    float v = hp[f];
    s1 += v * asp[f];
    s2 += v * adp[f];
  }
  asrc[i] = s1;
  adst[i] = s2;
}

// ---------------- edge pass 1: segment max (leaky-relu scores) -------------
__global__ void edge_max_kernel(const int* __restrict__ ei,
                                const float* __restrict__ asrc,
                                const float* __restrict__ adst,
                                unsigned* __restrict__ mkey, int H) {
  int i = blockIdx.x * blockDim.x + threadIdx.x;
  if (i >= NETOT * H) return;
  int e = i / H, hd = i - e * H;
  int src = (e < NEDGES) ? ei[e] : (e - NEDGES);
  int dst = (e < NEDGES) ? ei[NEDGES + e] : (e - NEDGES);
  float v = asrc[src * H + hd] + adst[dst * H + hd];
  v = v > 0.f ? v : 0.2f * v;                // leaky relu
  atomicMax(&mkey[dst * H + hd], f2o(v));
}

// ---------------- edge pass 2: exp + segment sum ---------------------------
__global__ void edge_exp_kernel(const int* __restrict__ ei,
                                const float* __restrict__ asrc,
                                const float* __restrict__ adst,
                                const unsigned* __restrict__ mkey,
                                float* __restrict__ exbuf,
                                float* __restrict__ ssum, int H) {
  int i = blockIdx.x * blockDim.x + threadIdx.x;
  if (i >= NETOT * H) return;
  int e = i / H, hd = i - e * H;
  int src = (e < NEDGES) ? ei[e] : (e - NEDGES);
  int dst = (e < NEDGES) ? ei[NEDGES + e] : (e - NEDGES);
  float v = asrc[src * H + hd] + adst[dst * H + hd];
  v = v > 0.f ? v : 0.2f * v;
  float ex = __expf(v - o2f(mkey[dst * H + hd]));
  exbuf[i] = ex;
  atomicAdd(&ssum[dst * H + hd], ex);
}

// ---------------- edge pass 3: weighted scatter-add of messages ------------
// One block per edge; one thread per 4 output features (blockDim == H*F/4).
__global__ void edge_aggr_kernel(const int* __restrict__ ei,
                                 const float* __restrict__ h,
                                 const float* __restrict__ exbuf,
                                 const float* __restrict__ ssum,
                                 float* __restrict__ out, int H, int F) {
  int e = blockIdx.x;
  int f4 = threadIdx.x << 2;    // 0 .. H*F-1, step 4 (within one head: F>=64)
  int hd = f4 / F;
  int src = (e < NEDGES) ? ei[e] : (e - NEDGES);
  int dst = (e < NEDGES) ? ei[NEDGES + e] : (e - NEDGES);
  float alpha = exbuf[(size_t)e * H + hd] / ssum[dst * H + hd];
  float4 hv = *(const float4*)(h + (size_t)src * H * F + f4);
  float* op = out + (size_t)dst * H * F + f4;
  atomicAdd(op + 0, hv.x * alpha);
  atomicAdd(op + 1, hv.y * alpha);
  atomicAdd(op + 2, hv.z * alpha);
  atomicAdd(op + 3, hv.w * alpha);
}

// ---------------- batch-norm statistics (per channel) ----------------------
__global__ void bn_stats_kernel(const float* __restrict__ x,
                                float* __restrict__ mu,
                                float* __restrict__ var, int n, int c) {
  __shared__ float ssum[256], ssq[256];
  int j = blockIdx.x, t = threadIdx.x;
  float s1 = 0.f, s2 = 0.f;
  for (int i = t; i < n; i += 256) {
    float v = x[(size_t)i * c + j];
    s1 += v; s2 += v * v;
  }
  ssum[t] = s1; ssq[t] = s2;
  __syncthreads();
  for (int off = 128; off > 0; off >>= 1) {
    if (t < off) { ssum[t] += ssum[t + off]; ssq[t] += ssq[t + off]; }
    __syncthreads();
  }
  if (t == 0) {
    float m = ssum[0] / (float)n;
    mu[j]  = m;
    var[j] = ssq[0] / (float)n - m * m;   // biased var, matches jnp.var
  }
}

// ---------------- BN + ELU -------------------------------------------------
__global__ void bn_elu_kernel(const float* __restrict__ x,
                              const float* __restrict__ mu,
                              const float* __restrict__ var,
                              const float* __restrict__ g,
                              const float* __restrict__ be,
                              float* __restrict__ y, int n, int c) {
  int i = blockIdx.x * blockDim.x + threadIdx.x;
  if (i >= n * c) return;
  int j = i % c;
  float v = (x[i] - mu[j]) * rsqrtf(var[j] + 1e-5f) * g[j] + be[j];
  y[i] = v > 0.f ? v : (__expf(v) - 1.f);
}

// ---------------- classifier head: 64 -> 32 (relu) -> 2 --------------------
__global__ void classifier_kernel(const float* __restrict__ h,
                                  const float* __restrict__ wc1,
                                  const float* __restrict__ bc1,
                                  const float* __restrict__ wc2,
                                  const float* __restrict__ bc2,
                                  float* __restrict__ out, int n) {
  __shared__ float w1s[64 * 32], w2s[64], b1s[32], b2s[2];
  for (int i = threadIdx.x; i < 64 * 32; i += blockDim.x) w1s[i] = wc1[i];
  for (int i = threadIdx.x; i < 64; i += blockDim.x)      w2s[i] = wc2[i];
  if (threadIdx.x < 32) b1s[threadIdx.x] = bc1[threadIdx.x];
  if (threadIdx.x < 2)  b2s[threadIdx.x] = bc2[threadIdx.x];
  __syncthreads();
  int node = blockIdx.x * blockDim.x + threadIdx.x;
  if (node >= n) return;
  float hreg[64];
#pragma unroll
  for (int f = 0; f < 64; ++f) hreg[f] = h[(size_t)node * 64 + f];
  float o0 = b2s[0], o1 = b2s[1];
  for (int j = 0; j < 32; ++j) {
    float acc = b1s[j];
#pragma unroll
    for (int f = 0; f < 64; ++f) acc += hreg[f] * w1s[f * 32 + j];
    acc = fmaxf(acc, 0.f);
    o0 += acc * w2s[j * 2 + 0];
    o1 += acc * w2s[j * 2 + 1];
  }
  out[node * 2 + 0] = o0;
  out[node * 2 + 1] = o1;
}

// ===========================================================================
// host side
// ===========================================================================
namespace {

struct Scratch {
  float*     bufH;    // [N,256] transformed features / layer output
  float*     bufO;    // [N,256] aggregated output
  _Float16*  x16;     // [N,256] f16 input to GEMM
  _Float16*  wpk;     // [256,256] f16 weights, fragment-packed
  float*     asrc;    // [N,H]
  float*     adst;    // [N,H]
  unsigned*  mkey;    // [N,H] ordered-float max keys
  float*     ssum;    // [N,H]
  float*     exbuf;   // [ETOT,H]
  float*     mu;      // [256]
  float*     var;     // [256]
};

inline void run_gat_layer(hipStream_t stream, const Scratch& S,
                          const float* in_f32, int K,
                          const float* W, const float* as, const float* ad,
                          const int* ei, int H, int F) {
  const int N = NNODES;
  const int Hout = H * F;

  // cast activations to f16; pack+cast weights into B-fragment order
  {
    int n4 = (N * K) >> 2;
    cast_f16_kernel<<<(n4 + 255) / 256, 256, 0, stream>>>(in_f32, S.x16, n4);
    int np = (K >> 5) * (Hout >> 4) * 32;
    pack_w_kernel<<<(np + 255) / 256, 256, 0, stream>>>(W, S.wpk, K, Hout);
  }
  // h = x16 @ W  (N x K) @ (K x Hout), 4 tiles (64 cols) per wave
  {
    dim3 grid(N / 16, Hout / 64);
    if (K == 128)
      gemm_wmma_kernel<128><<<grid, 32, 0, stream>>>(S.x16, S.wpk, S.bufH, Hout);
    else
      gemm_wmma_kernel<256><<<grid, 32, 0, stream>>>(S.x16, S.wpk, S.bufH, Hout);
  }
  // attention coefficients
  {
    int n = N * H;
    att_coef_kernel<<<(n + 255) / 256, 256, 0, stream>>>(
        S.bufH, as, ad, S.asrc, S.adst, N, H, F);
  }
  // init reductions
  hipMemsetAsync(S.mkey, 0, (size_t)N * H * sizeof(unsigned), stream);
  hipMemsetAsync(S.ssum, 0, (size_t)N * H * sizeof(float), stream);
  hipMemsetAsync(S.bufO, 0, (size_t)N * Hout * sizeof(float), stream);
  // edge passes
  {
    int n = NETOT * H;
    edge_max_kernel<<<(n + 255) / 256, 256, 0, stream>>>(
        ei, S.asrc, S.adst, S.mkey, H);
    edge_exp_kernel<<<(n + 255) / 256, 256, 0, stream>>>(
        ei, S.asrc, S.adst, S.mkey, S.exbuf, S.ssum, H);
    edge_aggr_kernel<<<NETOT, Hout / 4, 0, stream>>>(
        ei, S.bufH, S.exbuf, S.ssum, S.bufO, H, F);
  }
}

inline void run_bn_elu(hipStream_t stream, const Scratch& S,
                       const float* g, const float* be, int c) {
  bn_stats_kernel<<<c, 256, 0, stream>>>(S.bufO, S.mu, S.var, NNODES, c);
  int n = NNODES * c;
  bn_elu_kernel<<<(n + 255) / 256, 256, 0, stream>>>(
      S.bufO, S.mu, S.var, g, be, S.bufH, n / c, c);
}

} // namespace

extern "C" void kernel_launch(void* const* d_in, const int* in_sizes, int n_in,
                              void* d_out, int out_size, void* d_ws, size_t ws_size,
                              hipStream_t stream) {
  (void)in_sizes; (void)n_in; (void)out_size; (void)ws_size;

  const float* x   = (const float*)d_in[0];
  const int*   ei  = (const int*)d_in[1];
  const float* w0  = (const float*)d_in[2];
  const float* as0 = (const float*)d_in[4];
  const float* ad0 = (const float*)d_in[5];
  const float* g0  = (const float*)d_in[6];
  const float* be0 = (const float*)d_in[7];
  const float* w1  = (const float*)d_in[8];
  const float* as1 = (const float*)d_in[10];
  const float* ad1 = (const float*)d_in[11];
  const float* g1  = (const float*)d_in[12];
  const float* be1 = (const float*)d_in[13];
  const float* w2  = (const float*)d_in[14];
  const float* as2 = (const float*)d_in[16];
  const float* ad2 = (const float*)d_in[17];
  const float* g2  = (const float*)d_in[18];
  const float* be2 = (const float*)d_in[19];
  const float* wc1 = (const float*)d_in[20];
  const float* bc1 = (const float*)d_in[21];
  const float* wc2 = (const float*)d_in[22];
  const float* bc2 = (const float*)d_in[23];
  float* out = (float*)d_out;

  // --- carve workspace (256B-aligned) ---
  char* p = (char*)d_ws;
  auto carve = [&](size_t bytes) -> void* {
    void* r = (void*)p;
    p += (bytes + 255) & ~(size_t)255;
    return r;
  };
  Scratch S;
  const size_t P = (size_t)NNODES * 256;
  S.bufH  = (float*)carve(P * sizeof(float));
  S.bufO  = (float*)carve(P * sizeof(float));
  S.x16   = (_Float16*)carve(P * sizeof(_Float16));
  S.wpk   = (_Float16*)carve((size_t)256 * 256 * sizeof(_Float16));
  S.asrc  = (float*)carve((size_t)NNODES * 4 * sizeof(float));
  S.adst  = (float*)carve((size_t)NNODES * 4 * sizeof(float));
  S.mkey  = (unsigned*)carve((size_t)NNODES * 4 * sizeof(unsigned));
  S.ssum  = (float*)carve((size_t)NNODES * 4 * sizeof(float));
  S.exbuf = (float*)carve((size_t)NETOT * 4 * sizeof(float));
  S.mu    = (float*)carve(256 * sizeof(float));
  S.var   = (float*)carve(256 * sizeof(float));

  // --- layer 0: 128 -> 4x64 concat ---
  run_gat_layer(stream, S, x, 128, w0, as0, ad0, ei, /*H=*/4, /*F=*/64);
  run_bn_elu(stream, S, g0, be0, 256);            // -> bufH [N,256]

  // --- layer 1: 256 -> 4x64 concat ---
  run_gat_layer(stream, S, S.bufH, 256, w1, as1, ad1, ei, 4, 64);
  run_bn_elu(stream, S, g1, be1, 256);            // -> bufH [N,256]

  // --- layer 2: 256 -> 1x64 (mean over 1 head == identity) ---
  run_gat_layer(stream, S, S.bufH, 256, w2, as2, ad2, ei, 1, 64);
  run_bn_elu(stream, S, g2, be2, 64);             // -> bufH [N,64]

  // --- classifier head ---
  classifier_kernel<<<(NNODES + 127) / 128, 128, 0, stream>>>(
      S.bufH, wc1, bc1, wc2, bc2, out, NNODES);
}
